// WindowAttention_5076651344139
// MI455X (gfx1250) — matrix operand order repaired and checked by reading, not compiled, not run on hardware
//
#include <hip/hip_runtime.h>
#include <hip/hip_bf16.h>

typedef _Float16 half16 __attribute__((ext_vector_type(16)));
typedef _Float16 half8  __attribute__((ext_vector_type(8)));
typedef float    float8 __attribute__((ext_vector_type(8)));

#define NTOK 49
#define CDIM 256
#define NHH  8

// LDS layout (dynamic LDS base == 0, no static __shared__):
//   [0      .. 32767 ] xbuf  [64][256] f16   (reused as attn-out)
//   [32768  .. 98303 ] qk    per head Q[64][32],K[64][32] f16 (P overlays)
//   [98304  ..131071 ] vt    per head Vt[32][64] f16
//   [131072 ..136479 ] rpb   [169*8] f32
//   [136480 ..139551 ] qkv_b [768] f32
//   [139552 ..140575 ] proj_b[256] f32
#define RPB_OFF   131072u
#define QKVB_OFF  136480u
#define PROJB_OFF 139552u
#define SMEM_BYTES 140576

// ---------------- fragment loaders (wave32 WMMA; contiguous k order, same in A and B) ----------------
// A 16x32 f16: lane l (h=l>>4, m=l&15) holds row M=m; elem e -> K=kb+16h+e.
// Caller passes p = base + row*ldk + kb + 16h.
__device__ __forceinline__ half16 load_a16(const _Float16* p) {
  return *(const half16*)p;
}
// B 32x16 f16: lane l (h=l>>4, n=l&15) holds col N=n; elem e -> K=kb+16h+e.
// Source stored row-major as [N][K]; caller passes p = base + col*ldk + kb + 16h.
__device__ __forceinline__ half16 load_b16(const _Float16* p) {
  return *(const half16*)p;
}
__device__ __forceinline__ float8 wmma32(half16 a, half16 b, float8 c) {
  return __builtin_amdgcn_wmma_f32_16x16x32_f16(false, a, false, b, (short)0, c,
                                                false, false);
}
// async DMA one dword global -> LDS (per enabled lane); tracked by ASYNCcnt
__device__ __forceinline__ void async_copy_b32(unsigned lds_byte_off,
                                               const float* g) {
  asm volatile("global_load_async_to_lds_b32 %0, %1, off"
               :: "v"(lds_byte_off), "v"(g) : "memory");
}

// ---------------- kernel 0: transpose+convert weights to f16 [out][in] --------
__global__ __launch_bounds__(256) void prep_weights(
    const float* __restrict__ qkv_w, const float* __restrict__ proj_w,
    _Float16* __restrict__ qkvT, _Float16* __restrict__ projT) {
  int idx = blockIdx.x * 256 + threadIdx.x;
  if (idx < 768 * 256) {
    int o = idx >> 8, i = idx & 255;
    qkvT[o * 256 + i] = (_Float16)qkv_w[i * 768 + o];
  }
  if (idx < 256 * 256) {
    int o = idx >> 8, i = idx & 255;
    projT[o * 256 + i] = (_Float16)proj_w[i * 256 + o];
  }
}

// ---------------- fused window attention: one block per window ----------------
__global__ __launch_bounds__(256) void win_attn(
    const float* __restrict__ x, const _Float16* __restrict__ qkv_wT,
    const float* __restrict__ qkv_b, const float* __restrict__ rpb_table,
    const _Float16* __restrict__ proj_wT, const float* __restrict__ proj_b,
    float* __restrict__ out) {
  extern __shared__ char smem_raw[];
  _Float16* xbuf  = (_Float16*)smem_raw;        // [64][256]
  _Float16* qk    = xbuf + 64 * 256;            // per head Q,K (P overlays)
  _Float16* vt    = qk + 8 * 4096;              // per head Vt[32][64]
  float*    rpb   = (float*)(smem_raw + RPB_OFF);
  float*    biasq = (float*)(smem_raw + QKVB_OFF);
  float*    biasp = (float*)(smem_raw + PROJB_OFF);

  const int tid  = threadIdx.x;
  const int lane = tid & 31;
  const int wv   = tid >> 5;       // wave id 0..7
  const int h    = lane >> 4;      // lane half
  const int ln   = lane & 15;
  const int win  = blockIdx.x;
  const float scale = 0.17677669529663687f;  // 1/sqrt(32)
  const float8 zero = {};

  // Stage 0a: async DMA rpb + biases straight into LDS (no VGPR roundtrip)
  for (int i = tid; i < 169 * NHH; i += 256)
    async_copy_b32(RPB_OFF + 4u * i, rpb_table + i);
  for (int i = tid; i < 768; i += 256)
    async_copy_b32(QKVB_OFF + 4u * i, qkv_b + i);
  for (int i = tid; i < 256; i += 256)
    async_copy_b32(PROJB_OFF + 4u * i, proj_b + i);

  // Stage 0b: stage x window (pad rows to 64 with 0), f32 -> f16
  const float* xw = x + (size_t)win * NTOK * CDIM;
  for (int i = tid; i < 64 * 256; i += 256) {
    int r = i >> 8, c = i & 255;
    xbuf[i] = (_Float16)((r < NTOK) ? xw[r * CDIM + c] : 0.0f);
  }
  asm volatile("s_wait_asynccnt 0" ::: "memory");
  __syncthreads();

  // Stage 1: QKV GEMM 64x768, K=256. nt-major: each B fragment feeds 4 M-tiles
  for (int nt = wv; nt < 48; nt += 8) {
    if (nt + 8 < 48)
      __builtin_prefetch(qkv_wT + (size_t)((nt + 8) * 16 + ln) * 256 + 16 * h, 0, 0);
    const _Float16* bp = qkv_wT + (size_t)(nt * 16 + ln) * 256 + 16 * h;
    float8 acc[4] = {zero, zero, zero, zero};
#pragma unroll
    for (int k = 0; k < 8; ++k) {
      half16 b = load_b16(bp + k * 32);
#pragma unroll
      for (int mt = 0; mt < 4; ++mt) {
        half16 a = load_a16(xbuf + (mt * 16 + ln) * 256 + k * 32 + 16 * h);
        acc[mt] = wmma32(a, b, acc[mt]);
      }
    }
    int col   = nt * 16 + ln;     // 0..767 (which/head wave-uniform)
    float bb  = biasq[col];
    int which = col >> 8;         // 0=q 1=k 2=v
    int cc    = col & 255;
    int head  = cc >> 5;
    int d     = cc & 31;
    if (which < 2) {
      _Float16* dst = qk + head * 4096 + which * 2048;  // [64][32]
      float s = (which == 0) ? scale : 1.0f;
#pragma unroll
      for (int mt = 0; mt < 4; ++mt)
#pragma unroll
        for (int r = 0; r < 8; ++r) {
          int row = mt * 16 + r + 8 * h;
          dst[row * 32 + d] = (_Float16)((acc[mt][r] + bb) * s);
        }
    } else {
      _Float16* dst = vt + head * 2048;                 // Vt [32][64]
#pragma unroll
      for (int mt = 0; mt < 4; ++mt)
#pragma unroll
        for (int r = 0; r < 8; ++r) {
          int row = mt * 16 + r + 8 * h;
          dst[d * 64 + row] = (_Float16)(acc[mt][r] + bb);
        }
    }
  }
  __syncthreads();

  // Stage 2: attention, wave = head
  {
    const int head = wv;
    const _Float16* Qh = qk + head * 4096;
    const _Float16* Kh = Qh + 2048;
    _Float16* Ph = qk + head * 4096;   // P[64][64] overlays this head's Q+K
    const _Float16* Vh = vt + head * 2048;

    // S = (scale*Q) @ K^T : 4x4 tiles, single K=32 step (D==32)
    float8 S[4][4];
    half16 bK[4];
#pragma unroll
    for (int j = 0; j < 4; ++j)
      bK[j] = load_b16(Kh + (j * 16 + ln) * 32 + 16 * h);
#pragma unroll
    for (int i = 0; i < 4; ++i) {
      half16 a = load_a16(Qh + (i * 16 + ln) * 32 + 16 * h);
#pragma unroll
      for (int j = 0; j < 4; ++j) S[i][j] = wmma32(a, bK[j], zero);
    }

    // bias + mask + row softmax (rows live in 16-lane halves of the wave)
#pragma unroll
    for (int i = 0; i < 4; ++i) {
#pragma unroll
      for (int r = 0; r < 8; ++r) {
        int m  = i * 16 + r + 8 * h;          // query token (row)
        int mm = m < NTOK ? m : NTOK - 1;
        int my = mm / 7, mxc = mm - my * 7;
        float sv[4];
#pragma unroll
        for (int j = 0; j < 4; ++j) {
          int n = j * 16 + ln;                // key token (col)
          float v = -1e30f;
          if (n < NTOK) {
            int ny = n / 7, nxc = n - ny * 7;
            int rel = (my - ny + 6) * 13 + (mxc - nxc + 6);
            v = S[i][j][r] + rpb[rel * NHH + head];
          }
          sv[j] = v;
        }
        float rmax = fmaxf(fmaxf(sv[0], sv[1]), fmaxf(sv[2], sv[3]));
#pragma unroll
        for (int msk = 1; msk <= 8; msk <<= 1)
          rmax = fmaxf(rmax, __shfl_xor(rmax, msk, 32));
        float rsum = 0.f;
#pragma unroll
        for (int j = 0; j < 4; ++j) { sv[j] = __expf(sv[j] - rmax); rsum += sv[j]; }
#pragma unroll
        for (int msk = 1; msk <= 8; msk <<= 1)
          rsum += __shfl_xor(rsum, msk, 32);
        float inv = 1.0f / rsum;
#pragma unroll
        for (int j = 0; j < 4; ++j)
          Ph[m * 64 + j * 16 + ln] = (_Float16)(sv[j] * inv);
      }
    }

    // O = P @ V : A = P[64 x 64] (2 K-steps), B = Vt (contiguous in tokens)
    half16 bV[2][2];
#pragma unroll
    for (int ks = 0; ks < 2; ++ks)
#pragma unroll
      for (int dt = 0; dt < 2; ++dt)
        bV[ks][dt] = load_b16(Vh + (dt * 16 + ln) * 64 + ks * 32 + 16 * h);
#pragma unroll
    for (int i = 0; i < 4; ++i) {
      half16 a0 = load_a16(Ph + (i * 16 + ln) * 64 + 0 + 16 * h);
      half16 a1 = load_a16(Ph + (i * 16 + ln) * 64 + 32 + 16 * h);
#pragma unroll
      for (int dt = 0; dt < 2; ++dt) {
        float8 acc = wmma32(a0, bV[0][dt], zero);
        acc = wmma32(a1, bV[1][dt], acc);
#pragma unroll
        for (int r = 0; r < 8; ++r) {
          int row = i * 16 + r + 8 * h;
          xbuf[row * 256 + head * 32 + dt * 16 + ln] = (_Float16)acc[r];  // attn-out
        }
      }
    }
  }
  __syncthreads();

  // Stage 3: proj GEMM 64x256, K=256. nt-major, 4 M-tiles share each B fragment
  float* ow = out + (size_t)win * NTOK * CDIM;
  for (int nt = wv; nt < 16; nt += 8) {
    const _Float16* bp = proj_wT + (size_t)(nt * 16 + ln) * 256 + 16 * h;
    float8 acc[4] = {zero, zero, zero, zero};
#pragma unroll
    for (int k = 0; k < 8; ++k) {
      half16 b = load_b16(bp + k * 32);
#pragma unroll
      for (int mt = 0; mt < 4; ++mt) {
        half16 a = load_a16(xbuf + (mt * 16 + ln) * 256 + k * 32 + 16 * h);
        acc[mt] = wmma32(a, b, acc[mt]);
      }
    }
    int col  = nt * 16 + ln;
    float bb = biasp[col];
#pragma unroll
    for (int mt = 0; mt < 4; ++mt)
#pragma unroll
      for (int r = 0; r < 8; ++r) {
        int row = mt * 16 + r + 8 * h;
        if (row < NTOK) ow[row * CDIM + col] = acc[mt][r] + bb;
      }
  }
}

extern "C" void kernel_launch(void* const* d_in, const int* in_sizes, int n_in,
                              void* d_out, int out_size, void* d_ws, size_t ws_size,
                              hipStream_t stream) {
  (void)in_sizes; (void)n_in; (void)out_size; (void)ws_size;
  const float* x      = (const float*)d_in[0];
  const float* qkv_w  = (const float*)d_in[1];
  const float* qkv_b  = (const float*)d_in[2];
  const float* rpb    = (const float*)d_in[3];
  const float* proj_w = (const float*)d_in[4];
  const float* proj_b = (const float*)d_in[5];
  _Float16* qkvT  = (_Float16*)d_ws;                // [768][256] f16
  _Float16* projT = qkvT + 768 * 256;               // [256][256] f16

  prep_weights<<<768, 256, 0, stream>>>(qkv_w, proj_w, qkvT, projT);

  (void)hipFuncSetAttribute((const void*)win_attn,
                            hipFuncAttributeMaxDynamicSharedMemorySize,
                            SMEM_BYTES);
  win_attn<<<4096, 256, SMEM_BYTES, stream>>>(x, qkvT, qkv_b, rpb, projT,
                                              proj_b, (float*)d_out);
}
